// MambaBlock_12618613916216
// MI455X (gfx1250) — matrix operand order repaired and checked by reading, not compiled, not run on hardware
//
#include <hip/hip_runtime.h>
#include <math.h>

// Problem constants (match reference)
constexpr int kB   = 2;
constexpr int kL   = 1024;
constexpr int kDM  = 768;    // d_model
constexpr int kDI  = 1536;   // d_inner
constexpr int kDS  = 16;     // d_state
constexpr int kDTR = 48;     // dt_rank
constexpr int kDC  = 4;      // d_conv
constexpr float kLnEps = 1e-5f;
constexpr int kM = kB * kL;      // 2048 rows for all GEMMs
constexpr int kXD = 128;         // padded x_dbl row stride (dt|B|C|pad)
constexpr int kKdt = 64;         // padded K for the delta GEMM

typedef __attribute__((ext_vector_type(16))) _Float16 v16h;
typedef __attribute__((ext_vector_type(8)))  _Float16 v8h;
typedef __attribute__((ext_vector_type(8)))  float    v8f;

__device__ __forceinline__ float softplus_f(float x) {
  return x > 0.f ? x + log1pf(expf(-x)) : log1pf(expf(x));
}
__device__ __forceinline__ float silu_f(float x) {
  return x / (1.f + expf(-x));
}

// ---------------------------------------------------------------------------
// f16 fragment loaders (no bounds checks; all shapes exactly tiled/padded).
// A (16x32 MxK): lane m = m0 + (lane&15); the lane's 16 halves are two
//   contiguous 8-half runs at k0 + 8*(lane>=16) and +16 further.
// B (32x16 KxN) from W[N x ldw] row-major (B[k][n] = W[n][k]): lane n = n0 +
//   (lane&15); two contiguous 8-half runs at k0 + 8*(lane>=16) and +16 further.
// C/D: v8f, VGPR r: M = r + 8*(lane>=16), N = lane&15.
// ---------------------------------------------------------------------------
__device__ __forceinline__ v16h load_a_frag(const _Float16* __restrict__ A,
                                            int lda, int m0, int k0, int lane) {
  const _Float16* p =
      A + (size_t)(m0 + (lane & 15)) * lda + k0 + ((lane >> 4) << 3);
  const v8h lo = *(const v8h*)p;         // K offsets +0..7   -> b128
  const v8h hi = *(const v8h*)(p + 16);  // K offsets +16..23 -> b128
  return __builtin_shufflevector(lo, hi, 0, 1, 2, 3, 4, 5, 6, 7, 8, 9, 10, 11,
                                 12, 13, 14, 15);
}

__device__ __forceinline__ v16h load_b_frag(const _Float16* __restrict__ W,
                                            int ldw, int n0, int k0, int lane) {
  const _Float16* p =
      W + (size_t)(n0 + (lane & 15)) * ldw + k0 + ((lane >> 4) << 3);
  const v8h lo = *(const v8h*)p;  // K offsets +0..7   -> b128
  const v8h hi = *(const v8h*)(p + 16);  // K offsets +16..23 -> b128
  return __builtin_shufflevector(lo, hi, 0, 1, 2, 3, 4, 5, 6, 7, 8, 9, 10, 11,
                                 12, 13, 14, 15);
}

// ---------------------------------------------------------------------------
// GEMM: C[M,N] = A[M,K](f16) * W[N,K](f16)^T, f32 accumulate. No guards:
// M % 64 == 0, N % 128 == 0 (grid-exact), K % 32 == 0, all loads in-bounds.
// Software-pipelined: iteration i+1's fragments are loaded before iteration
// i's 4 WMMAs consume theirs, so WMMAs wait on a non-empty load pipe instead
// of draining it (relaxed s_wait_loadcnt).
// EPI 0: store f32 C.
// EPI 1: store f32 softplus(C + bias[n])                (delta path)
// EPI 2: store f32 C and f16 C (dual)                   (x_proj path)
// Block = 256 threads = 8 waves tiled 4(M) x 2(N); each wave: 16x64 strip.
// ---------------------------------------------------------------------------
template <int EPI>
__global__ __launch_bounds__(256) void gemm_wmma(
    const _Float16* __restrict__ A, const _Float16* __restrict__ W,
    const float* __restrict__ bias, float* __restrict__ C,
    _Float16* __restrict__ C2, int K, int lda, int ldw, int ldc) {
  const int lane = threadIdx.x & 31;
  const int wave = threadIdx.x >> 5;  // 0..7
  const int m0 = blockIdx.y * 64 + (wave >> 1) * 16;
  const int nbase = blockIdx.x * 128 + (wave & 1) * 64;

  const v8f vzero = {};
  v8f acc0 = vzero, acc1 = vzero, acc2 = vzero, acc3 = vzero;

  // Prologue: fragments for k0 = 0.
  v16h a  = load_a_frag(A, lda, m0, 0, lane);
  v16h b0 = load_b_frag(W, ldw, nbase + 0,  0, lane);
  v16h b1 = load_b_frag(W, ldw, nbase + 16, 0, lane);
  v16h b2 = load_b_frag(W, ldw, nbase + 32, 0, lane);
  v16h b3 = load_b_frag(W, ldw, nbase + 48, 0, lane);

  for (int k0 = 0; k0 < K; k0 += 32) {
    // Preload next k-tile (last iteration harmlessly re-reads tile 0).
    int kn = k0 + 32;
    if (kn >= K) kn = 0;
    const v16h an  = load_a_frag(A, lda, m0, kn, lane);
    const v16h bn0 = load_b_frag(W, ldw, nbase + 0,  kn, lane);
    const v16h bn1 = load_b_frag(W, ldw, nbase + 16, kn, lane);
    const v16h bn2 = load_b_frag(W, ldw, nbase + 32, kn, lane);
    const v16h bn3 = load_b_frag(W, ldw, nbase + 48, kn, lane);

    acc0 = __builtin_amdgcn_wmma_f32_16x16x32_f16(false, a, false, b0,
                                                  (short)0, acc0, false, false);
    acc1 = __builtin_amdgcn_wmma_f32_16x16x32_f16(false, a, false, b1,
                                                  (short)0, acc1, false, false);
    acc2 = __builtin_amdgcn_wmma_f32_16x16x32_f16(false, a, false, b2,
                                                  (short)0, acc2, false, false);
    acc3 = __builtin_amdgcn_wmma_f32_16x16x32_f16(false, a, false, b3,
                                                  (short)0, acc3, false, false);

    a = an; b0 = bn0; b1 = bn1; b2 = bn2; b3 = bn3;
  }

  const int cn = lane & 15;
  const int moff = (lane >> 4) << 3;  // 0 or 8
  v8f accs[4] = {acc0, acc1, acc2, acc3};
#pragma unroll
  for (int t = 0; t < 4; ++t) {
    const int n = nbase + t * 16 + cn;
    const float bn = (EPI == 1) ? bias[n] : 0.f;
#pragma unroll
    for (int r = 0; r < 8; ++r) {
      const int m = m0 + moff + r;
      float v = accs[t][r];
      if (EPI == 1) v = softplus_f(v + bn);
      C[(size_t)m * ldc + n] = v;
      if (EPI == 2) C2[(size_t)m * ldc + n] = (_Float16)v;
    }
  }
}

// ---------------------------------------------------------------------------
// Convert f32 [Nsrc x Ksrc] -> f16 [Npad x Kpad], zero-filling the padding.
// ---------------------------------------------------------------------------
__global__ __launch_bounds__(256) void cvt_pad_f16(
    const float* __restrict__ src, _Float16* __restrict__ dst, int Nsrc,
    int Ksrc, int Npad, int Kpad) {
  const int idx = blockIdx.x * 256 + threadIdx.x;
  if (idx >= Npad * Kpad) return;
  const int k = idx % Kpad;
  const int n = idx / Kpad;
  const float v = (n < Nsrc && k < Ksrc) ? src[(size_t)n * Ksrc + k] : 0.f;
  dst[idx] = (_Float16)v;
}

// ---------------------------------------------------------------------------
// Depthwise causal conv (width 4) over time + bias + SiLU.
// u = xz[:, :, 0:DI] (xz row stride 2*DI). Writes u_c (f32, for scan) and
// u_ch (f16, A operand of x_proj).
// ---------------------------------------------------------------------------
__global__ __launch_bounds__(256) void conv_silu_kernel(
    const float* __restrict__ xz, const float* __restrict__ conv_w,
    const float* __restrict__ conv_b, float* __restrict__ u_c,
    _Float16* __restrict__ u_ch) {
  const int idx = blockIdx.x * blockDim.x + threadIdx.x;  // over B*L*DI
  if (idx >= kB * kL * kDI) return;
  const int d = idx % kDI;
  const int l = (idx / kDI) % kL;
  const int b = idx / (kDI * kL);
  float acc = conv_b[d];
#pragma unroll
  for (int k = 0; k < kDC; ++k) {
    const int t = l - (kDC - 1) + k;
    if (t >= 0)
      acc += xz[((size_t)(b * kL + t)) * (2 * kDI) + d] * conv_w[d * kDC + k];
  }
  const float s = silu_f(acc);
  u_c[idx] = s;
  u_ch[idx] = (_Float16)s;
}

// ---------------------------------------------------------------------------
// Selective scan. One thread per (b, d) channel; 16 states in registers.
// delta/u_c/z loads coalesced across d; B/C loads lane-uniform (broadcast).
// Fuses D*u skip + silu(z) gate; emits f16 (feeds out_proj only).
// ---------------------------------------------------------------------------
__global__ __launch_bounds__(256) void scan_kernel(
    const float* __restrict__ u_c, const float* __restrict__ delta,
    const float* __restrict__ xdbl, const float* __restrict__ xz,
    const float* __restrict__ A_log, const float* __restrict__ D_param,
    _Float16* __restrict__ yh) {
  constexpr int kChanBlocks = kDI / 256;  // 6
  const int b = blockIdx.x / kChanBlocks;
  const int d = (blockIdx.x % kChanBlocks) * 256 + threadIdx.x;

  float Ad[kDS], h[kDS];
#pragma unroll
  for (int n = 0; n < kDS; ++n) {
    Ad[n] = -expf(A_log[d * kDS + n]);
    h[n] = 0.f;
  }
  const float Dd = D_param[d];

  for (int t = 0; t < kL; ++t) {
    const size_t row = (size_t)(b * kL + t);
    const float dv = delta[row * kDI + d];
    const float uv = u_c[row * kDI + d];
    const float zv = xz[row * (2 * kDI) + kDI + d];
    const float* bc = xdbl + row * kXD + kDTR;  // B (16) then C (16)
    const float du = dv * uv;
    float yv = 0.f;
#pragma unroll
    for (int n = 0; n < kDS; ++n) {
      h[n] = h[n] * expf(dv * Ad[n]) + du * bc[n];
      yv = fmaf(h[n], bc[kDS + n], yv);
    }
    yv = fmaf(uv, Dd, yv);
    yv *= silu_f(zv);
    yh[row * kDI + d] = (_Float16)yv;
  }
}

// ---------------------------------------------------------------------------
// LayerNorm over last dim (768). One 256-thread block per row.
// ---------------------------------------------------------------------------
__global__ __launch_bounds__(256) void layernorm_kernel(
    const float* __restrict__ in, const float* __restrict__ w,
    const float* __restrict__ bprm, float* __restrict__ out) {
  const int row = blockIdx.x;  // B*L rows
  const float* x = in + (size_t)row * kDM;
  float v[3];
  float s = 0.f, ss = 0.f;
#pragma unroll
  for (int i = 0; i < 3; ++i) {
    v[i] = x[threadIdx.x + i * 256];
    s += v[i];
    ss += v[i] * v[i];
  }
  __shared__ float s1[256], s2[256];
  s1[threadIdx.x] = s;
  s2[threadIdx.x] = ss;
  __syncthreads();
  for (int off = 128; off > 0; off >>= 1) {
    if (threadIdx.x < off) {
      s1[threadIdx.x] += s1[threadIdx.x + off];
      s2[threadIdx.x] += s2[threadIdx.x + off];
    }
    __syncthreads();
  }
  const float mu = s1[0] * (1.f / kDM);
  const float var = s2[0] * (1.f / kDM) - mu * mu;
  const float rstd = rsqrtf(var + kLnEps);
#pragma unroll
  for (int i = 0; i < 3; ++i) {
    const int c = threadIdx.x + i * 256;
    out[(size_t)row * kDM + c] = (v[i] - mu) * rstd * w[c] + bprm[c];
  }
}

// ---------------------------------------------------------------------------
// Launch: cvt weights -> in_proj -> conv -> x_proj -> delta -> scan ->
//         out_proj -> LayerNorm
// ---------------------------------------------------------------------------
extern "C" void kernel_launch(void* const* d_in, const int* in_sizes, int n_in,
                              void* d_out, int out_size, void* d_ws, size_t ws_size,
                              hipStream_t stream) {
  (void)in_sizes; (void)n_in; (void)out_size; (void)ws_size;
  const float* x       = (const float*)d_in[0];
  const float* W_in    = (const float*)d_in[1];
  const float* conv_w  = (const float*)d_in[2];
  const float* conv_b  = (const float*)d_in[3];
  const float* W_x     = (const float*)d_in[4];
  const float* W_dt    = (const float*)d_in[5];
  const float* b_dt    = (const float*)d_in[6];
  const float* A_log   = (const float*)d_in[7];
  const float* D_param = (const float*)d_in[8];
  const float* W_out   = (const float*)d_in[9];
  const float* ln_w    = (const float*)d_in[10];
  const float* ln_b    = (const float*)d_in[11];
  float* out = (float*)d_out;

  // Workspace carve-up (f32 buffers first, then f16; all 16B-aligned).
  float* wsf = (float*)d_ws;
  float* xz    = wsf; wsf += (size_t)kM * 2 * kDI;  // 2048 x 3072
  float* u_c   = wsf; wsf += (size_t)kM * kDI;      // 2048 x 1536
  float* xdbl  = wsf; wsf += (size_t)kM * kXD;      // 2048 x 128 (dt|B|C|0)
  float* delta = wsf; wsf += (size_t)kM * kDI;      // 2048 x 1536
  float* pre   = wsf; wsf += (size_t)kM * kDM;      // 2048 x 768

  _Float16* wsh = (_Float16*)wsf;
  _Float16* xh      = wsh; wsh += (size_t)kM * kDM;       // 2048 x 768
  _Float16* Winh    = wsh; wsh += (size_t)2 * kDI * kDM;  // 3072 x 768
  _Float16* Wxh     = wsh; wsh += (size_t)kXD * kDI;      // 128 x 1536 (pad)
  _Float16* Wdth    = wsh; wsh += (size_t)kDI * kKdt;     // 1536 x 64  (pad)
  _Float16* Wouth   = wsh; wsh += (size_t)kDM * kDI;      // 768 x 1536
  _Float16* u_ch    = wsh; wsh += (size_t)kM * kDI;       // 2048 x 1536
  _Float16* xdbl_h  = wsh; wsh += (size_t)kM * kXD;       // 2048 x 128
  _Float16* yh      = wsh; wsh += (size_t)kM * kDI;       // 2048 x 1536

  const dim3 blk(256);
  auto grid1 = [](int n) { return dim3((n + 255) / 256); };

  // 0) f32 -> f16 conversions (+ zero padding where needed)
  cvt_pad_f16<<<grid1(kM * kDM), blk, 0, stream>>>(x, xh, kM, kDM, kM, kDM);
  cvt_pad_f16<<<grid1(2 * kDI * kDM), blk, 0, stream>>>(W_in, Winh, 2 * kDI,
                                                        kDM, 2 * kDI, kDM);
  cvt_pad_f16<<<grid1(kXD * kDI), blk, 0, stream>>>(W_x, Wxh, kDTR + 2 * kDS,
                                                    kDI, kXD, kDI);
  cvt_pad_f16<<<grid1(kDI * kKdt), blk, 0, stream>>>(W_dt, Wdth, kDI, kDTR,
                                                     kDI, kKdt);
  cvt_pad_f16<<<grid1(kDM * kDI), blk, 0, stream>>>(W_out, Wouth, kDM, kDI,
                                                    kDM, kDI);

  // 1) in_proj: xz = x @ W_in^T   (M=2048, N=3072, K=768)
  {
    dim3 grid(2 * kDI / 128, kM / 64);
    gemm_wmma<0><<<grid, blk, 0, stream>>>(xh, Winh, nullptr, xz, nullptr,
                                           kDM, kDM, kDM, 2 * kDI);
  }
  // 2) depthwise causal conv + SiLU (f32 + f16 outputs)
  conv_silu_kernel<<<grid1(kM * kDI), blk, 0, stream>>>(xz, conv_w, conv_b,
                                                        u_c, u_ch);
  // 3) x_proj: x_dbl = u_c @ W_x^T  (N padded 80->128, K=1536, dual store)
  {
    dim3 grid(kXD / 128, kM / 64);
    gemm_wmma<2><<<grid, blk, 0, stream>>>(u_ch, Wxh, nullptr, xdbl, xdbl_h,
                                           kDI, kDI, kDI, kXD);
  }
  // 4) delta = softplus(dt @ W_dt^T + b_dt)  (N=1536, K padded 48->64)
  {
    dim3 grid(kDI / 128, kM / 64);
    gemm_wmma<1><<<grid, blk, 0, stream>>>(xdbl_h, Wdth, b_dt, delta, nullptr,
                                           kKdt, kXD, kKdt, kDI);
  }
  // 5) selective scan (+ D*u skip + silu(z) gate) -> f16
  scan_kernel<<<kB * (kDI / 256), blk, 0, stream>>>(u_c, delta, xdbl, xz,
                                                    A_log, D_param, yh);
  // 6) out_proj: pre = y @ W_out^T  (N=768, K=1536)
  {
    dim3 grid(kDM / 128, kM / 64);
    gemm_wmma<0><<<grid, blk, 0, stream>>>(yh, Wouth, nullptr, pre, nullptr,
                                           kDI, kDI, kDI, kDM);
  }
  // 7) LayerNorm -> d_out
  layernorm_kernel<<<kM, blk, 0, stream>>>(pre, ln_w, ln_b, out);
}
